// LocalMHA_13030930776303
// MI455X (gfx1250) — matrix operand order repaired and checked by reading, not compiled, not run on hardware
//
#include <hip/hip_runtime.h>

// ---------------- problem constants ----------------
#define DIM    1024
#define T_LEN  2048
#define B_SZ   2
#define HEADS  16
#define DH     64
#define WIN    32
#define NROWS  (B_SZ * T_LEN)   // 4096 token rows
#define N_QKV  (3 * DIM)        // 3072

// ---------------- vector types for WMMA ----------------
typedef __attribute__((ext_vector_type(16))) __bf16       v16bf;
typedef __attribute__((ext_vector_type(8)))  float        v8f;
typedef __attribute__((ext_vector_type(4)))  unsigned int v4ui;

struct FragRaw { v4ui lo, hi; };   // 32 bytes = 16 bf16

__device__ __forceinline__ unsigned short f32_to_bf16(float f) {
  union { float f; unsigned u; } v; v.f = f;
  unsigned r = v.u + 0x7FFFu + ((v.u >> 16) & 1u);   // round-to-nearest-even
  return (unsigned short)(r >> 16);
}

// A-matrix 16x32 bf16 fragment (ISA 7.12.2): lane L<16 -> row M=L, halves are
// K = {k8+0..7, 16+k8+0..7} with k8 = (L>>4)*8. p points at (row, k_base+k8).
__device__ __forceinline__ v16bf load_frag_a(const unsigned short* p) {
  FragRaw fr;
  fr.lo = *(const v4ui*)(p);        // K chunk 0..7
  fr.hi = *(const v4ui*)(p + 16);   // K chunk 16..23
  return __builtin_bit_cast(v16bf, fr);
}

// B-matrix 32x16 bf16 fragment: lane holds col N = lane&15 and K =
// {k8+0..7, 16+k8+0..7} with k8 = (lane>>4)*8 (same K order as A).
__device__ __forceinline__ v16bf load_frag_b(const unsigned short* p) {
  FragRaw fr;
  fr.lo = *(const v4ui*)(p);
  fr.hi = *(const v4ui*)(p + 16);
  return __builtin_bit_cast(v16bf, fr);
}

__device__ __forceinline__ v8f wmma_bf16(v16bf a, v16bf b, v8f c) {
  return __builtin_amdgcn_wmma_f32_16x16x32_bf16(
      /*neg_a=*/false, a, /*neg_b=*/false, b,
      /*c_mod=*/(short)0, c, /*reuse_a=*/false, /*reuse_b=*/false);
}

// ---------------- 1) weight transpose + f32 -> bf16 ----------------
// dst[n*1024 + c] = bf16(src[c*cols + n]);  rows of src are always K=1024.
__global__ __launch_bounds__(256) void wcast_kernel(const float* __restrict__ src,
                                                    unsigned short* __restrict__ dst,
                                                    int cols) {
  int id = blockIdx.x * 256 + threadIdx.x;
  int c = id & 1023;
  int n = id >> 10;
  if (n < cols) dst[id] = f32_to_bf16(src[(size_t)c * cols + n]);
}

// ---------------- 2) LayerNorm over channels, x is [B, C, T] ----------------
// One block handles 16 consecutive t positions; LDS tile [1024][16] padded to
// stride 17 floats to avoid bank conflicts. Output H[B*T, C] in bf16.
__global__ __launch_bounds__(256) void ln_kernel(const float* __restrict__ x,
                                                 const float* __restrict__ lnw,
                                                 const float* __restrict__ lnb,
                                                 unsigned short* __restrict__ H) {
  __shared__ float tile[1024 * 17];
  const int b     = blockIdx.x >> 7;            // T/16 = 128 blocks per batch
  const int tbase = (blockIdx.x & 127) * 16;
  const int tid   = threadIdx.x;
  const int tt_ld = tid & 15;
  const int cc_ld = tid >> 4;                   // 16 channels per pass

  const float* xb = x + (size_t)b * DIM * T_LEN;
#pragma unroll 4
  for (int pass = 0; pass < 64; ++pass) {
    int c = pass * 16 + cc_ld;
    tile[c * 17 + tt_ld] = xb[(size_t)c * T_LEN + tbase + tt_ld];
  }
  __syncthreads();

  const int wave = tid >> 5;
  const int lane = tid & 31;
  for (int sub = 0; sub < 2; ++sub) {
    const int tt = wave * 2 + sub;
    float sum = 0.f, sq = 0.f;
#pragma unroll
    for (int i = 0; i < 32; ++i) {
      float v = tile[(lane + 32 * i) * 17 + tt];
      sum += v; sq += v * v;
    }
#pragma unroll
    for (int off = 16; off > 0; off >>= 1) {
      sum += __shfl_xor(sum, off, 32);
      sq  += __shfl_xor(sq,  off, 32);
    }
    const float mu  = sum * (1.f / 1024.f);
    const float var = sq * (1.f / 1024.f) - mu * mu;
    const float rs  = rsqrtf(var + 1e-5f);
    unsigned short* hrow = H + ((size_t)(b * T_LEN + tbase + tt)) * DIM;
#pragma unroll
    for (int i = 0; i < 32; ++i) {
      int c = lane + 32 * i;
      float o = (tile[c * 17 + tt] - mu) * rs * lnw[c] + lnb[c];
      hrow[c] = f32_to_bf16(o);
    }
  }
}

// ---------------- 3) QKV GEMM: C[4096,3072] = H[4096,1024] @ WqkvT^T ----------
// 8 waves/block; wave tile 32x32 (2x2 WMMA), block tile 128(M) x 64(N).
__global__ __launch_bounds__(256) void qkv_gemm(const unsigned short* __restrict__ A,
                                                const unsigned short* __restrict__ Bt,
                                                float* __restrict__ C) {
  const int K = DIM, N = N_QKV;
  const int lane = threadIdx.x & 31;
  const int wave = threadIdx.x >> 5;
  const int m0 = blockIdx.x * 128 + (wave & 3) * 32;
  const int n0 = blockIdx.y * 64  + (wave >> 2) * 32;
  const int r16   = lane & 15;
  const int a_off = (lane >> 4) * 8;
  const int b_off = (lane >> 4) * 8;

  const unsigned short* a0 = A  + (size_t)(m0 + r16)      * K + a_off;
  const unsigned short* a1 = a0 + (size_t)16 * K;
  const unsigned short* b0 = Bt + (size_t)(n0 + r16)      * K + b_off;
  const unsigned short* b1 = b0 + (size_t)16 * K;

  v8f acc00 = {}, acc01 = {}, acc10 = {}, acc11 = {};
  for (int kk = 0; kk < K; kk += 32) {
    if (kk + 32 < K) {                 // global_prefetch_b8 of next K tile
      __builtin_prefetch(a0 + kk + 32, 0, 1);
      __builtin_prefetch(b0 + kk + 32, 0, 1);
    }
    v16bf fa0 = load_frag_a(a0 + kk);
    v16bf fa1 = load_frag_a(a1 + kk);
    v16bf fb0 = load_frag_b(b0 + kk);
    v16bf fb1 = load_frag_b(b1 + kk);
    acc00 = wmma_bf16(fa0, fb0, acc00);
    acc01 = wmma_bf16(fa0, fb1, acc01);
    acc10 = wmma_bf16(fa1, fb0, acc10);
    acc11 = wmma_bf16(fa1, fb1, acc11);
  }
  const int mhi = (lane >> 4) * 8;
  const int n   = n0 + (lane & 15);
#pragma unroll
  for (int r = 0; r < 8; ++r) {
    int m = m0 + mhi + r;
    C[(size_t)m * N + n]             = acc00[r];
    C[(size_t)m * N + n + 16]        = acc01[r];
    C[(size_t)(m + 16) * N + n]      = acc10[r];
    C[(size_t)(m + 16) * N + n + 16] = acc11[r];
  }
}

// ---------------- 4) RoPE on q and k (in place) ----------------
__global__ __launch_bounds__(256) void rope_kernel(float* __restrict__ qkv) {
  int gid = blockIdx.x * 256 + threadIdx.x;     // NROWS * 1024 threads
  int row = gid >> 10;
  int s   = gid & 1023;
  int mat = s >> 9;        // 0 = q, 1 = k
  int rem = s & 511;
  int hh  = rem >> 5;
  int i   = rem & 31;      // rotation pair index
  int t   = row & (T_LEN - 1);
  float* base = qkv + (size_t)row * N_QKV + mat * DIM + hh * DH;
  float f  = (float)t * __powf(10000.f, -(float)i * (1.f / 32.f));
  float sn, cs;
  __sincosf(f, &sn, &cs);
  float a  = base[i];
  float bb = base[i + 32];
  base[i]      = a * cs - bb * sn;
  base[i + 32] = bb * cs + a * sn;
}

// ---------------- 5) windowed causal attention: one wave32 per query ----------
// Window 32 == wave width: lane j owns key position t-31+j.
__global__ __launch_bounds__(256) void attn_kernel(const float* __restrict__ qkv,
                                                   unsigned short* __restrict__ Abuf) {
  const int lane = threadIdx.x & 31;
  const int wid  = blockIdx.x * 8 + (threadIdx.x >> 5);
  const int b = wid >> 15;          // / (HEADS*T_LEN)
  const int h = (wid >> 11) & 15;
  const int t = wid & (T_LEN - 1);

  const float* qrow = qkv + ((size_t)(b * T_LEN + t)) * N_QKV + h * DH;
  const int kp      = t - (WIN - 1) + lane;
  const bool valid  = kp >= 0;
  const float* krow = qkv + ((size_t)(b * T_LEN + (valid ? kp : 0))) * N_QKV + DIM + h * DH;

  float score = 0.f;
#pragma unroll
  for (int d = 0; d < DH; d += 4) {
    float4 qv = *(const float4*)(qrow + d);
    float4 kv = *(const float4*)(krow + d);
    score += qv.x * kv.x + qv.y * kv.y + qv.z * kv.z + qv.w * kv.w;
  }
  score = valid ? score * 0.125f : -1e30f;      // 1/sqrt(64)

  float m = score;
#pragma unroll
  for (int off = 16; off > 0; off >>= 1) m = fmaxf(m, __shfl_xor(m, off, 32));
  float p = valid ? __expf(score - m) : 0.f;
  float s = p;
#pragma unroll
  for (int off = 16; off > 0; off >>= 1) s += __shfl_xor(s, off, 32);
  p /= s;

  // AV: lane owns output dims (2*lane, 2*lane+1); v reads are lane-coalesced.
  const int d0 = lane * 2;
  float acc0 = 0.f, acc1 = 0.f;
  const float* vbase = qkv + (size_t)b * T_LEN * N_QKV + 2 * DIM + h * DH + d0;
#pragma unroll 4
  for (int j = 0; j < WIN; ++j) {
    float pj  = __shfl(p, j, 32);
    int   kpj = t - (WIN - 1) + j;
    if (kpj >= 0) {
      float2 vv = *(const float2*)(vbase + (size_t)kpj * N_QKV);
      acc0 += pj * vv.x;
      acc1 += pj * vv.y;
    }
  }
  unsigned short* arow = Abuf + ((size_t)(b * T_LEN + t)) * DIM + h * DH + d0;
  arow[0] = f32_to_bf16(acc0);
  arow[1] = f32_to_bf16(acc1);
}

// ---------------- 6) out GEMM + transpose back to [B,C,T] + residual ---------
__global__ __launch_bounds__(256) void out_gemm(const unsigned short* __restrict__ A,
                                                const unsigned short* __restrict__ Bt,
                                                const float* __restrict__ X,
                                                float* __restrict__ out) {
  const int K = DIM;
  const int lane = threadIdx.x & 31;
  const int wave = threadIdx.x >> 5;
  const int m0 = blockIdx.x * 128 + (wave & 3) * 32;
  const int n0 = blockIdx.y * 64  + (wave >> 2) * 32;
  const int r16   = lane & 15;
  const int a_off = (lane >> 4) * 8;
  const int b_off = (lane >> 4) * 8;

  const unsigned short* a0 = A  + (size_t)(m0 + r16) * K + a_off;
  const unsigned short* a1 = a0 + (size_t)16 * K;
  const unsigned short* b0 = Bt + (size_t)(n0 + r16) * K + b_off;
  const unsigned short* b1 = b0 + (size_t)16 * K;

  v8f acc00 = {}, acc01 = {}, acc10 = {}, acc11 = {};
  for (int kk = 0; kk < K; kk += 32) {
    if (kk + 32 < K) {
      __builtin_prefetch(a0 + kk + 32, 0, 1);
      __builtin_prefetch(b0 + kk + 32, 0, 1);
    }
    v16bf fa0 = load_frag_a(a0 + kk);
    v16bf fa1 = load_frag_a(a1 + kk);
    v16bf fb0 = load_frag_b(b0 + kk);
    v16bf fb1 = load_frag_b(b1 + kk);
    acc00 = wmma_bf16(fa0, fb0, acc00);
    acc01 = wmma_bf16(fa0, fb1, acc01);
    acc10 = wmma_bf16(fa1, fb0, acc10);
    acc11 = wmma_bf16(fa1, fb1, acc11);
  }
  const int mhi = (lane >> 4) * 8;
  const int n   = n0 + (lane & 15);
#pragma unroll
  for (int r = 0; r < 8; ++r) {
    int m = m0 + mhi + r;                 // m = b*T + t; wave tile stays in one b
    int b = m >> 11;
    int t = m & (T_LEN - 1);
    size_t o00 = ((size_t)(b * DIM + n)) * T_LEN + t;
    size_t o01 = ((size_t)(b * DIM + n + 16)) * T_LEN + t;
    out[o00]      = acc00[r] + X[o00];
    out[o01]      = acc01[r] + X[o01];
    out[o00 + 16] = acc10[r] + X[o00 + 16];   // m+16 -> t+16, same b
    out[o01 + 16] = acc11[r] + X[o01 + 16];
  }
}

// ---------------- launcher ----------------
extern "C" void kernel_launch(void* const* d_in, const int* in_sizes, int n_in,
                              void* d_out, int out_size, void* d_ws, size_t ws_size,
                              hipStream_t stream) {
  const float* x     = (const float*)d_in[0];
  const float* ln_w  = (const float*)d_in[1];
  const float* ln_b  = (const float*)d_in[2];
  const float* w_qkv = (const float*)d_in[3];
  const float* w_out = (const float*)d_in[4];
  float*       out   = (float*)d_out;

  // workspace layout (all 256B-aligned by construction):
  unsigned short* H     = (unsigned short*)d_ws;                 //  8 MB bf16 [4096,1024]
  unsigned short* WqkvT = H     + (size_t)NROWS * DIM;           //  6 MB bf16 [3072,1024]
  unsigned short* WoutT = WqkvT + (size_t)N_QKV * DIM;           //  2 MB bf16 [1024,1024]
  float*          qkvB  = (float*)(WoutT + (size_t)DIM * DIM);   // 48 MB f32  [4096,3072]
  unsigned short* Abuf  = (unsigned short*)(qkvB + (size_t)NROWS * N_QKV); // 8 MB bf16

  // 1) cast + transpose weights to bf16 [N][K]
  wcast_kernel<<<(N_QKV * DIM) / 256, 256, 0, stream>>>(w_qkv, WqkvT, N_QKV);
  wcast_kernel<<<(DIM * DIM) / 256, 256, 0, stream>>>(w_out, WoutT, DIM);

  // 2) LayerNorm -> H (bf16, [B*T, C])
  ln_kernel<<<NROWS / 16, 256, 0, stream>>>(x, ln_w, ln_b, H);

  // 3) QKV projection (WMMA bf16)
  qkv_gemm<<<dim3(NROWS / 128, N_QKV / 64), 256, 0, stream>>>(H, WqkvT, qkvB);

  // 4) RoPE on q, k
  rope_kernel<<<(NROWS * 1024) / 256, 256, 0, stream>>>(qkvB);

  // 5) windowed attention -> Abuf (bf16, [B*T, C])
  attn_kernel<<<(NROWS * HEADS) / 8, 256, 0, stream>>>(qkvB, Abuf);

  // 6) output projection + residual + transpose to [B, C, T]
  out_gemm<<<dim3(NROWS / 128, DIM / 64), 256, 0, stream>>>(Abuf, WoutT, x, out);
}
